// CARP_26938034881182
// MI455X (gfx1250) — matrix operand order repaired and checked by reading, not compiled, not run on hardware
//
#include <hip/hip_runtime.h>
#include <hip/hip_bf16.h>

// ---------------------------------------------------------------------------
// Problem constants (from reference)
// ---------------------------------------------------------------------------
#define B_     8
#define NPG_   2048
#define N0_    (B_ * NPG_)     // 16384
#define E_     131072
#define ND_    64
#define ED_    16
#define H_     4
#define OC_    128
#define HOC_   (H_ * OC_)      // 512
#define GIC_   256
#define MIC_   8192            // rows of ic
#define EPS_   1e-5f

#define CDIV(a, b) (((a) + (b) - 1) / (b))

typedef __attribute__((ext_vector_type(16))) _Float16 v16h;
typedef __attribute__((ext_vector_type(8)))  _Float16 v8h;
typedef __attribute__((ext_vector_type(8)))  float    v8f;

// ---------------------------------------------------------------------------
// Order-preserving float <-> uint for atomic segment max
// ---------------------------------------------------------------------------
__device__ __forceinline__ unsigned f2u_ord(float f) {
    unsigned u = __float_as_uint(f);
    return (u & 0x80000000u) ? ~u : (u | 0x80000000u);
}
__device__ __forceinline__ float u2f_ord(unsigned k) {
    return (k & 0x80000000u) ? __uint_as_float(k & 0x7fffffffu)
                             : __uint_as_float(~k);
}

// ---------------------------------------------------------------------------
// Generic WMMA GEMM:  C[M,N] = act(gather(A)[M,K] @ W[K,N] + bias[N])
// A row-major f32 (optionally row-gathered), W row-major f32.
// Block = 8 waves; each wave computes a 16x64 C strip as 4 x (16x16) tiles
// with v_wmma_f32_16x16x32_f16 (A fragment reused across the 4 tiles).
// B tile (32x64) staged in LDS *transposed* [n][k] so each lane's fragment
// is two contiguous, 16B-aligned ds_load_b128 reads.
// Main K loop is branchless (K multiple of 32); guarded tail handles K%32.
// ---------------------------------------------------------------------------
__global__ __launch_bounds__(256) void wmma_gemm_kernel(
    const float* __restrict__ A, const int* __restrict__ rowIdx, int lda,
    const float* __restrict__ W, const float* __restrict__ bias,
    float* __restrict__ C, int ldc, int M, int K, int N, int doRelu)
{
    __shared__ __attribute__((aligned(16))) _Float16 Bs[64][32];  // [n][k]

    const int lane = threadIdx.x & 31;
    const int wave = threadIdx.x >> 5;
    const int m0   = (blockIdx.x * 8 + wave) * 16;
    const int n0   = blockIdx.y * 64;
    const int half = lane >> 4;   // 0: lanes 0-15, 1: lanes 16-31
    const int mr   = lane & 15;

    // Per-lane A row (ISA: 16-bit A 16x32 — both lane halves map M=0..15).
    // Clamp out-of-range rows instead of branching: they only feed C rows
    // that are never stored.
    int row = m0 + mr;
    if (row >= M) row = M - 1;
    const long arow = rowIdx ? (long)rowIdx[row] : (long)row;
    const float* __restrict__ Aptr = A + arow * (long)lda;

    v8f acc[4] = {};

    const int Kmain = K & ~31;
    int k0 = 0;
    for (; k0 < Kmain; k0 += 32) {
        // Stage B[k0..k0+31][n0..n0+63] -> Bs[n][k] (coalesced reads over n)
        for (int idx = threadIdx.x; idx < 32 * 64; idx += 256) {
            int kk = idx >> 6, nn = idx & 63;
            Bs[nn][kk] = (_Float16)W[(long)(k0 + kk) * N + (n0 + nn)];
        }
        if (threadIdx.x == 0 && k0 + 32 < K)
            __builtin_prefetch(&W[(long)(k0 + 32) * N + n0], 0, 3); // global_prefetch_b8
        __syncthreads();

        // A fragment: half 0 -> K {0..15}; half 1 -> K {16..31}
        v16h a;
        const float* __restrict__ ap = Aptr + k0 + half * 16;
        #pragma unroll
        for (int i = 0; i < 8; ++i) {
            a[i]     = (_Float16)ap[i];
            a[i + 8] = (_Float16)ap[8 + i];
        }
        // 4 WMMAs sharing the A fragment
        #pragma unroll
        for (int nt = 0; nt < 4; ++nt) {
            union { v16h v; v8h h[2]; } bu;
            const _Float16* bp = &Bs[nt * 16 + mr][half * 16];
            bu.h[0] = *(const v8h*)(bp);
            bu.h[1] = *(const v8h*)(bp + 8);
            acc[nt] = __builtin_amdgcn_wmma_f32_16x16x32_f16(
                false, a, false, bu.v, (short)0, acc[nt], false, false);
        }
        __syncthreads();
    }

    if (k0 < K) {  // guarded tail (K % 32 != 0; also whole GEMM when K < 32)
        for (int idx = threadIdx.x; idx < 32 * 64; idx += 256) {
            int kk = idx >> 6, nn = idx & 63;
            float bv = (k0 + kk < K) ? W[(long)(k0 + kk) * N + (n0 + nn)] : 0.f;
            Bs[nn][kk] = (_Float16)bv;
        }
        __syncthreads();
        v16h a;
        #pragma unroll
        for (int i = 0; i < 8; ++i) {
            int ka  = k0 + half * 16 + i;
            int ka2 = k0 + half * 16 + 8 + i;
            a[i]     = (_Float16)((ka  < K) ? Aptr[ka]  : 0.f);
            a[i + 8] = (_Float16)((ka2 < K) ? Aptr[ka2] : 0.f);
        }
        #pragma unroll
        for (int nt = 0; nt < 4; ++nt) {
            union { v16h v; v8h h[2]; } bu;
            const _Float16* bp = &Bs[nt * 16 + mr][half * 16];
            bu.h[0] = *(const v8h*)(bp);
            bu.h[1] = *(const v8h*)(bp + 8);
            acc[nt] = __builtin_amdgcn_wmma_f32_16x16x32_f16(
                false, a, false, bu.v, (short)0, acc[nt], false, false);
        }
        __syncthreads();
    }

    // Epilogue: lane holds column n; acc[nt][r] -> row m0 + r + 8*half
    #pragma unroll
    for (int nt = 0; nt < 4; ++nt) {
        int n = n0 + nt * 16 + mr;
        float bv = bias ? bias[n] : 0.f;
        #pragma unroll
        for (int r = 0; r < 8; ++r) {
            int m = m0 + r + 8 * half;
            if (m < M) {
                float o = acc[nt][r] + bv;
                if (doRelu) o = fmaxf(o, 0.f);
                C[(long)m * ldc + n] = o;
            }
        }
    }
}

// ---------------------------------------------------------------------------
// Edge attention kernels (transformer_conv)
// ---------------------------------------------------------------------------
__global__ void edge_alpha_kernel(const float* __restrict__ q,
                                  const float* __restrict__ k,
                                  const float* __restrict__ e,
                                  const int* __restrict__ ei,
                                  const int* __restrict__ valid,
                                  float* __restrict__ alpha,
                                  int E, float scale)
{
    int t = blockIdx.x * blockDim.x + threadIdx.x;
    if (t >= E * H_) return;
    int eidx = t >> 2, h = t & 3;
    if (valid && !valid[eidx]) { alpha[t] = -1e30f; return; }
    int src = ei[eidx], dst = ei[E + eidx];
    const float4* qp = (const float4*)(q + (long)dst * HOC_ + h * OC_);
    const float4* kp = (const float4*)(k + (long)src * HOC_ + h * OC_);
    const float4* ep = (const float4*)(e + (long)eidx * HOC_ + h * OC_);
    float s = 0.f;
    for (int c = 0; c < OC_ / 4; ++c) {
        float4 qv = qp[c], kv = kp[c], ev = ep[c];
        s += qv.x * (kv.x + ev.x) + qv.y * (kv.y + ev.y)
           + qv.z * (kv.z + ev.z) + qv.w * (kv.w + ev.w);
    }
    alpha[t] = s * scale;
}

__global__ void edge_segmax_kernel(const float* __restrict__ alpha,
                                   const int* __restrict__ ei,
                                   const int* __restrict__ valid,
                                   unsigned* __restrict__ amaxk, int E)
{
    int t = blockIdx.x * blockDim.x + threadIdx.x;
    if (t >= E * H_) return;
    int eidx = t >> 2, h = t & 3;
    if (valid && !valid[eidx]) return;
    int dst = ei[E + eidx];
    atomicMax(&amaxk[dst * H_ + h], f2u_ord(alpha[t]));
}

__global__ void edge_ex_kernel(float* __restrict__ alpha,  // in: alpha, out: ex
                               const int* __restrict__ ei,
                               const int* __restrict__ valid,
                               const unsigned* __restrict__ amaxk,
                               float* __restrict__ den, int E)
{
    int t = blockIdx.x * blockDim.x + threadIdx.x;
    if (t >= E * H_) return;
    int eidx = t >> 2, h = t & 3;
    if (valid && !valid[eidx]) { alpha[t] = 0.f; return; }
    int dst = ei[E + eidx];
    float am = fmaxf(u2f_ord(amaxk[dst * H_ + h]), -1e29f);
    float ex = expf(alpha[t] - am);
    alpha[t] = ex;
    atomicAdd(&den[dst * H_ + h], ex);
}

// one thread per (edge, 4-channel chunk): float4 loads + 4 atomic adds
__global__ void edge_msg_kernel(const float* __restrict__ v,
                                const float* __restrict__ e,
                                const float* __restrict__ ex,
                                const float* __restrict__ den,
                                const int* __restrict__ ei,
                                const int* __restrict__ valid,
                                float* __restrict__ agg, long total4, int E)
{
    long t = (long)blockIdx.x * blockDim.x + threadIdx.x;
    if (t >= total4) return;
    int eidx = (int)(t >> 7);
    int j = (int)(t & 127) * 4;       // channel base within 512
    int h = j >> 7;
    if (valid && !valid[eidx]) return;
    int src = ei[eidx], dst = ei[E + eidx];
    float a = ex[eidx * H_ + h] / fmaxf(den[dst * H_ + h], 1e-16f);
    float4 vv = *(const float4*)(v + (long)src * HOC_ + j);
    float4 ee = *(const float4*)(e + (long)eidx * HOC_ + j);
    float* ap = agg + (long)dst * HOC_ + j;
    atomicAdd(ap + 0, (vv.x + ee.x) * a);
    atomicAdd(ap + 1, (vv.y + ee.y) * a);
    atomicAdd(ap + 2, (vv.z + ee.z) * a);
    atomicAdd(ap + 3, (vv.w + ee.w) * a);
}

__global__ void head_mean_skip_kernel(const float* __restrict__ agg,
                                      const float* __restrict__ skip,
                                      float* __restrict__ out, int N)
{
    int t = blockIdx.x * blockDim.x + threadIdx.x;
    if (t >= N * OC_) return;
    int n = t >> 7, c = t & 127;
    const float* ap = agg + (long)n * HOC_ + c;
    float s = ap[0] + ap[128] + ap[256] + ap[384];
    out[t] = 0.25f * s + skip[t];
}

// ---------------------------------------------------------------------------
// Graph norm (column stats over N, then apply + relu)
// ---------------------------------------------------------------------------
__global__ void gn_stats_kernel(const float* __restrict__ x, int N,
                                const float* __restrict__ msArr,
                                float* __restrict__ mu, float* __restrict__ var)
{
    int c = blockIdx.x;  // 0..127
    float s1 = 0.f, s2 = 0.f;
    for (int n = threadIdx.x; n < N; n += blockDim.x) {
        float v = x[(long)n * OC_ + c];
        s1 += v; s2 += v * v;
    }
    __shared__ float r1[256], r2[256];
    r1[threadIdx.x] = s1; r2[threadIdx.x] = s2;
    __syncthreads();
    for (int s = 128; s > 0; s >>= 1) {
        if ((int)threadIdx.x < s) { r1[threadIdx.x] += r1[threadIdx.x + s];
                                    r2[threadIdx.x] += r2[threadIdx.x + s]; }
        __syncthreads();
    }
    if (threadIdx.x == 0) {
        float m = r1[0] / (float)N;
        float ms = msArr[c];
        float vv = r2[0] / (float)N - 2.f * m * ms * m + (m * ms) * (m * ms);
        mu[c] = m; var[c] = vv;
    }
}

__global__ void gn_apply_relu_kernel(const float* __restrict__ x,
                                     const float* __restrict__ w,
                                     const float* __restrict__ b,
                                     const float* __restrict__ msArr,
                                     const float* __restrict__ mu,
                                     const float* __restrict__ var,
                                     float* __restrict__ y, int N)
{
    int t = blockIdx.x * blockDim.x + threadIdx.x;
    if (t >= N * OC_) return;
    int c = t & 127;
    float o = x[t] - mu[c] * msArr[c];
    float r = w[c] * o / sqrtf(var[c] + EPS_) + b[c];
    y[t] = fmaxf(r, 0.f);
}

// ---------------------------------------------------------------------------
// Top-k pooling
// ---------------------------------------------------------------------------
__global__ void score_kernel(const float* __restrict__ x,
                             const float* __restrict__ w,
                             float* __restrict__ s, int N)
{
    int n = blockIdx.x * blockDim.x + threadIdx.x;
    if (n >= N) return;
    float nw = 0.f, d = 0.f;
    const float* xp = x + (long)n * OC_;
    for (int c = 0; c < OC_; ++c) { nw += w[c] * w[c]; d += xp[c] * w[c]; }
    s[n] = tanhf(d / sqrtf(nw));
}

// one block per batch group; bitonic sort (desc by value, tie: idx asc)
__global__ __launch_bounds__(1024) void topk_kernel(
    const float* __restrict__ score, int npg, int k,
    int* __restrict__ perm, float* __restrict__ vals)
{
    __shared__ float sv[2048];
    __shared__ int   si[2048];
    int b = blockIdx.x;
    for (int i = threadIdx.x; i < npg; i += blockDim.x) {
        sv[i] = score[b * npg + i]; si[i] = i;
    }
    __syncthreads();
    for (int size = 2; size <= npg; size <<= 1) {
        for (int stride = size >> 1; stride > 0; stride >>= 1) {
            for (int i = threadIdx.x; i < npg; i += blockDim.x) {
                int j = i ^ stride;
                if (j > i) {
                    bool desc = ((i & size) == 0);
                    float vi = sv[i], vj = sv[j];
                    int ii = si[i], ij = si[j];
                    bool iBefore = (vi > vj) || (vi == vj && ii < ij);
                    bool doSwap = desc ? !iBefore : iBefore;
                    if (doSwap) { sv[i] = vj; sv[j] = vi; si[i] = ij; si[j] = ii; }
                }
            }
            __syncthreads();
        }
    }
    for (int j = threadIdx.x; j < k; j += blockDim.x) {
        perm[b * k + j] = b * npg + si[j];
        vals[b * k + j] = sv[j];
    }
}

__global__ void ni_scatter_kernel(int* __restrict__ ni,
                                  const int* __restrict__ perm, int n)
{
    int i = blockIdx.x * blockDim.x + threadIdx.x;
    if (i < n) ni[perm[i]] = i;
}

__global__ void gather_rows_scale_kernel(const float* __restrict__ x,
                                         const int* __restrict__ perm,
                                         const float* __restrict__ vals,
                                         float* __restrict__ y, int nRows)
{
    int t = blockIdx.x * blockDim.x + threadIdx.x;
    if (t >= nRows * (OC_ / 4)) return;
    int i = t >> 5, c4 = (t & 31) * 4;
    float4 xv = *(const float4*)(x + (long)perm[i] * OC_ + c4);
    float s = vals[i];
    float4 yv = make_float4(xv.x * s, xv.y * s, xv.z * s, xv.w * s);
    *(float4*)(y + (long)i * OC_ + c4) = yv;
}

__global__ void gather_f_kernel(const float* __restrict__ src,
                                const int* __restrict__ perm,
                                float* __restrict__ dst, int n)
{
    int i = blockIdx.x * blockDim.x + threadIdx.x;
    if (i < n) dst[i] = src[perm[i]];
}
__global__ void gather_i_kernel(const int* __restrict__ src,
                                const int* __restrict__ perm,
                                int* __restrict__ dst, int n)
{
    int i = blockIdx.x * blockDim.x + threadIdx.x;
    if (i < n) dst[i] = src[perm[i]];
}

__global__ void remap_ei_kernel(const int* __restrict__ ei_in,
                                const int* __restrict__ valid_in,
                                const int* __restrict__ ni,
                                int* __restrict__ ei_out,
                                int* __restrict__ valid_out, int E)
{
    int e = blockIdx.x * blockDim.x + threadIdx.x;
    if (e >= E) return;
    int s = ni[ei_in[e]], d = ni[ei_in[E + e]];
    int v = (valid_in ? valid_in[e] : 1) && (s >= 0) && (d >= 0);
    valid_out[e] = v;
    ei_out[e]     = v ? s : 0;
    ei_out[E + e] = v ? d : 0;
}

__global__ void em_valid_kernel(const int* __restrict__ ei,
                                const float* __restrict__ pk,
                                int* __restrict__ valid, int E)
{
    int e = blockIdx.x * blockDim.x + threadIdx.x;
    if (e >= E) return;
    if (!valid[e]) return;
    int s = ei[e], d = ei[E + e];
    bool em = ((pk[s] != 0.f) != (pk[d] != 0.f));
    if (!em) valid[e] = 0;
}

__global__ void ic_remap_init_kernel(const int* __restrict__ ic,
                                     const int* __restrict__ ni,
                                     int* __restrict__ ic0,
                                     int* __restrict__ icv, int M)
{
    int m = blockIdx.x * blockDim.x + threadIdx.x;
    if (m >= M) return;
    int i0 = ni[ic[m * 2]];
    int v = (i0 >= 0);
    icv[m] = v;
    ic0[m] = v ? i0 : 0;
}

__global__ void ic_remap_next_kernel(int* __restrict__ ic0,
                                     int* __restrict__ icv,
                                     const int* __restrict__ ni, int M)
{
    int m = blockIdx.x * blockDim.x + threadIdx.x;
    if (m >= M) return;
    int i0 = ni[ic0[m]];
    int v = icv[m] && (i0 >= 0);
    icv[m] = v;
    ic0[m] = v ? i0 : 0;
}

// ---------------------------------------------------------------------------
// nfc = concat(nf, pk, pri)  (4096 x 130)
// ---------------------------------------------------------------------------
__global__ void nfc_kernel(const float* __restrict__ nf,
                           const float* __restrict__ pk,
                           const float* __restrict__ pri,
                           float* __restrict__ nfc, int N)
{
    int t = blockIdx.x * blockDim.x + threadIdx.x;
    if (t >= N * 130) return;
    int n = t / 130, c = t % 130;
    float v = (c < OC_) ? nf[(long)n * OC_ + c] : (c == OC_ ? pk[n] : pri[n]);
    nfc[t] = v;
}

// ---------------------------------------------------------------------------
// Attention pooling
// ---------------------------------------------------------------------------
__global__ void gate_kernel(const float* __restrict__ x,
                            const int* __restrict__ rowIdx, int ld,
                            const float* __restrict__ Wg,
                            const float* __restrict__ bg,
                            float* __restrict__ gate, int M, int K)
{
    int m = blockIdx.x * blockDim.x + threadIdx.x;
    if (m >= M) return;
    long r = rowIdx ? (long)rowIdx[m] : (long)m;
    const float* xp = x + r * ld;
    float s = bg[0];
    for (int c = 0; c < K; ++c) s += xp[c] * Wg[c];
    gate[m] = fmaxf(s, 0.f);
}

__global__ void pool_segmax_kernel(const float* __restrict__ gate,
                                   const int* __restrict__ seg, int segStride,
                                   const int* __restrict__ validm,
                                   unsigned* __restrict__ keys, int M)
{
    int m = blockIdx.x * blockDim.x + threadIdx.x;
    if (m >= M) return;
    if (validm && !validm[m]) return;
    atomicMax(&keys[seg[m * segStride]], f2u_ord(gate[m]));
}

__global__ void pool_ex_kernel(float* __restrict__ gate,  // in gate, out ex
                               const int* __restrict__ seg, int segStride,
                               const int* __restrict__ validm,
                               const unsigned* __restrict__ keys,
                               float* __restrict__ den, int M)
{
    int m = blockIdx.x * blockDim.x + threadIdx.x;
    if (m >= M) return;
    if (validm && !validm[m]) { gate[m] = 0.f; return; }
    int g = seg[m * segStride];
    float am = fmaxf(u2f_ord(keys[g]), -1e29f);
    float ex = expf(gate[m] - am);
    gate[m] = ex;
    atomicAdd(&den[g], ex);
}

__global__ void pool_agg_kernel(const float* __restrict__ ex,
                                const float* __restrict__ den,
                                const int* __restrict__ seg, int segStride,
                                const int* __restrict__ validm,
                                const float* __restrict__ mproj,
                                float* __restrict__ out, int M)
{
    long t = (long)blockIdx.x * blockDim.x + threadIdx.x;
    if (t >= (long)M * OC_) return;
    int m = (int)(t >> 7), c = (int)(t & 127);
    if (validm && !validm[m]) return;
    int g = seg[m * segStride];
    float a = ex[m] / fmaxf(den[g], 1e-16f);
    atomicAdd(&out[g * OC_ + c], a * mproj[(long)m * OC_ + c]);
}

__global__ void head_kernel(const float* __restrict__ pooled,
                            const float* __restrict__ W,
                            const float* __restrict__ b,
                            float* __restrict__ dst,
                            int G, int O, int rowStride, int colOff)
{
    int t = blockIdx.x * blockDim.x + threadIdx.x;
    if (t >= G * O) return;
    int g = t / O, o = t % O;
    const float* pp = pooled + (long)g * OC_;
    float s = b[o];
    for (int c = 0; c < OC_; ++c) s += pp[c] * W[c * O + o];
    dst[g * rowStride + colOff + o] = s;
}

__global__ void fill_u32_kernel(unsigned* __restrict__ p, unsigned v, int n)
{
    int i = blockIdx.x * blockDim.x + threadIdx.x;
    if (i < n) p[i] = v;
}

// ---------------------------------------------------------------------------
// Host-side orchestration
// ---------------------------------------------------------------------------
struct ConvP { const float *Wq,*bq,*Wk,*bk,*Wv,*bv,*We,*be,*Ws,*bs; };
struct GnP   { const float *w,*b,*ms; };
struct PoolP { const float *Wg,*bg,*Wm,*bm; };

static void launch_gemm(const float* A, const int* rowIdx, int lda,
                        const float* W, const float* bias,
                        float* C, int ldc, int M, int K, int N, int relu,
                        hipStream_t s)
{
    dim3 g(CDIV(M, 128), CDIV(N, 64));   // N must be a multiple of 64 (512/128 used)
    wmma_gemm_kernel<<<g, 256, 0, s>>>(A, rowIdx, lda, W, bias, C, ldc, M, K, N, relu);
}

static unsigned host_f2u_ord(float f) {
    unsigned u;
    __builtin_memcpy(&u, &f, 4);
    return (u & 0x80000000u) ? ~u : (u | 0x80000000u);
}

static void run_conv(const float* xin, int lda, int K, int Nn,
                     const int* ei, const int* valid,
                     const float* ef, const ConvP& cp, const GnP& gp,
                     float* q, float* k, float* v, float* ebuf,
                     float* alpha, unsigned* amaxk, float* den, float* agg,
                     float* skip, float* hbuf, float* mu, float* var,
                     float* xout, hipStream_t s)
{
    // Projections (WMMA)
    launch_gemm(xin, nullptr, lda, cp.Wq, cp.bq, q,    HOC_, Nn, K,   HOC_, 0, s);
    launch_gemm(xin, nullptr, lda, cp.Wk, cp.bk, k,    HOC_, Nn, K,   HOC_, 0, s);
    launch_gemm(xin, nullptr, lda, cp.Wv, cp.bv, v,    HOC_, Nn, K,   HOC_, 0, s);
    launch_gemm(ef,  nullptr, ED_, cp.We, cp.be, ebuf, HOC_, E_, ED_, HOC_, 0, s);
    launch_gemm(xin, nullptr, lda, cp.Ws, cp.bs, skip, OC_,  Nn, K,   OC_,  0, s);

    const float scale = 0.088388347648318447f;  // 1/sqrt(128)
    int eth = E_ * H_;
    edge_alpha_kernel<<<CDIV(eth, 256), 256, 0, s>>>(q, k, ebuf, ei, valid, alpha, E_, scale);
    fill_u32_kernel<<<CDIV(Nn * H_, 256), 256, 0, s>>>(amaxk, host_f2u_ord(-1e30f), Nn * H_);
    edge_segmax_kernel<<<CDIV(eth, 256), 256, 0, s>>>(alpha, ei, valid, amaxk, E_);
    hipMemsetAsync(den, 0, (size_t)Nn * H_ * sizeof(float), s);
    edge_ex_kernel<<<CDIV(eth, 256), 256, 0, s>>>(alpha, ei, valid, amaxk, den, E_);
    hipMemsetAsync(agg, 0, (size_t)Nn * HOC_ * sizeof(float), s);
    long total4 = (long)E_ * (HOC_ / 4);
    edge_msg_kernel<<<(int)CDIV(total4, 256), 256, 0, s>>>(v, ebuf, alpha, den, ei, valid, agg, total4, E_);
    head_mean_skip_kernel<<<CDIV(Nn * OC_, 256), 256, 0, s>>>(agg, skip, hbuf, Nn);

    // GraphNorm + ReLU
    gn_stats_kernel<<<OC_, 256, 0, s>>>(hbuf, Nn, gp.ms, mu, var);
    gn_apply_relu_kernel<<<CDIV(Nn * OC_, 256), 256, 0, s>>>(hbuf, gp.w, gp.b, gp.ms, mu, var, xout, Nn);
}

static void run_attn_pool(const float* x, int ld, int Kdim,
                          const int* rowIdx, int Mrows,
                          const int* seg, int segStride,
                          const int* validm, int G, const PoolP& pp,
                          float* gate, unsigned* keys, float* den,
                          float* mproj, float* pooled, hipStream_t s)
{
    gate_kernel<<<CDIV(Mrows, 256), 256, 0, s>>>(x, rowIdx, ld, pp.Wg, pp.bg, gate, Mrows, Kdim);
    fill_u32_kernel<<<CDIV(G, 256), 256, 0, s>>>(keys, host_f2u_ord(-1e30f), G);
    pool_segmax_kernel<<<CDIV(Mrows, 256), 256, 0, s>>>(gate, seg, segStride, validm, keys, Mrows);
    hipMemsetAsync(den, 0, (size_t)G * sizeof(float), s);
    pool_ex_kernel<<<CDIV(Mrows, 256), 256, 0, s>>>(gate, seg, segStride, validm, keys, den, Mrows);
    launch_gemm(x, rowIdx, ld, pp.Wm, pp.bm, mproj, OC_, Mrows, Kdim, OC_, 1, s);
    hipMemsetAsync(pooled, 0, (size_t)G * OC_ * sizeof(float), s);
    long total = (long)Mrows * OC_;
    pool_agg_kernel<<<(int)CDIV(total, 256), 256, 0, s>>>(gate, den, seg, segStride, validm, mproj, pooled, Mrows);
}

extern "C" void kernel_launch(void* const* d_in, const int* in_sizes, int n_in,
                              void* d_out, int out_size, void* d_ws, size_t ws_size,
                              hipStream_t stream)
{
    (void)in_sizes; (void)n_in; (void)out_size; (void)ws_size;

    // ---- inputs (setup_inputs dict order, nested structures flattened) ----
    const float* nf   = (const float*)d_in[0];
    const float* ef   = (const float*)d_in[1];
    const float* pk   = (const float*)d_in[2];
    const float* pri  = (const float*)d_in[3];
    const int*   ei   = (const int*)d_in[4];
    const int*   ic   = (const int*)d_in[5];
    const int*   batch= (const int*)d_in[6];

    int p = 7;
    ConvP cp[3];
    for (int l = 0; l < 3; ++l) {
        cp[l].Wq = (const float*)d_in[p + 0]; cp[l].bq = (const float*)d_in[p + 1];
        cp[l].Wk = (const float*)d_in[p + 2]; cp[l].bk = (const float*)d_in[p + 3];
        cp[l].Wv = (const float*)d_in[p + 4]; cp[l].bv = (const float*)d_in[p + 5];
        cp[l].We = (const float*)d_in[p + 6]; cp[l].be = (const float*)d_in[p + 7];
        cp[l].Ws = (const float*)d_in[p + 8]; cp[l].bs = (const float*)d_in[p + 9];
        p += 10;
    }
    GnP gp[3];
    for (int l = 0; l < 3; ++l) {
        gp[l].w = (const float*)d_in[p]; gp[l].b = (const float*)d_in[p + 1];
        gp[l].ms = (const float*)d_in[p + 2];
        p += 3;
    }
    const float* tkw0 = (const float*)d_in[p++];
    const float* tkw1 = (const float*)d_in[p++];
    PoolP pp[3];  // iface, gfold, giface
    for (int i = 0; i < 3; ++i) {
        pp[i].Wg = (const float*)d_in[p]; pp[i].bg = (const float*)d_in[p + 1];
        pp[i].Wm = (const float*)d_in[p + 2]; pp[i].bm = (const float*)d_in[p + 3];
        p += 4;
    }
    const float* Wi  = (const float*)d_in[p + 0];
    const float* bi  = (const float*)d_in[p + 1];
    const float* Wf  = (const float*)d_in[p + 2];
    const float* bf  = (const float*)d_in[p + 3];
    const float* Wgi = (const float*)d_in[p + 4];
    const float* bgi = (const float*)d_in[p + 5];

    // ---- workspace bump allocator ----
    char* wsb = (char*)d_ws;
    size_t off = 0;
    auto alloc = [&](size_t bytes) -> void* {
        void* q = wsb + off;
        off += (bytes + 255) & ~(size_t)255;
        return q;
    };
    float* qb    = (float*)alloc((size_t)N0_ * HOC_ * 4);
    float* kb    = (float*)alloc((size_t)N0_ * HOC_ * 4);
    float* vb    = (float*)alloc((size_t)N0_ * HOC_ * 4);
    float* ebuf  = (float*)alloc((size_t)E_ * HOC_ * 4);
    float* alpha = (float*)alloc((size_t)E_ * H_ * 4);
    unsigned* amaxk = (unsigned*)alloc((size_t)N0_ * H_ * 4);
    float* den   = (float*)alloc((size_t)N0_ * H_ * 4);
    float* agg   = (float*)alloc((size_t)N0_ * HOC_ * 4);
    float* skip  = (float*)alloc((size_t)N0_ * OC_ * 4);
    float* hbuf  = (float*)alloc((size_t)N0_ * OC_ * 4);
    float* x0    = (float*)alloc((size_t)N0_ * OC_ * 4);
    float* x1    = (float*)alloc((size_t)N0_ * OC_ * 4);
    float* x2    = (float*)alloc((size_t)8192 * OC_ * 4);
    float* x3    = (float*)alloc((size_t)8192 * OC_ * 4);
    float* x4    = (float*)alloc((size_t)4096 * OC_ * 4);
    float* mu    = (float*)alloc(OC_ * 4);
    float* var   = (float*)alloc(OC_ * 4);
    float* score = (float*)alloc((size_t)N0_ * 4);
    int*   perm  = (int*)alloc((size_t)8192 * 4);
    float* vals  = (float*)alloc((size_t)8192 * 4);
    int*   ni    = (int*)alloc((size_t)N0_ * 4);
    int*   ei1   = (int*)alloc((size_t)2 * E_ * 4);
    int*   valid1= (int*)alloc((size_t)E_ * 4);
    float* pkB   = (float*)alloc(8192 * 4);
    float* priB  = (float*)alloc(8192 * 4);
    int*   batchB= (int*)alloc(8192 * 4);
    float* pkC   = (float*)alloc(4096 * 4);
    float* priC  = (float*)alloc(4096 * 4);
    int*   batchC= (int*)alloc(4096 * 4);
    int*   ic0   = (int*)alloc((size_t)MIC_ * 4);
    int*   icv   = (int*)alloc((size_t)MIC_ * 4);
    float* nfc   = (float*)alloc((size_t)4096 * 130 * 4);
    float* gate  = (float*)alloc((size_t)MIC_ * 4);
    unsigned* pkeys = (unsigned*)alloc((size_t)GIC_ * 4);
    float* pden  = (float*)alloc((size_t)GIC_ * 4);
    float* mproj = (float*)alloc((size_t)MIC_ * OC_ * 4);
    float* pooledI = (float*)alloc((size_t)GIC_ * OC_ * 4);
    float* pooledF = (float*)alloc((size_t)B_ * OC_ * 4);
    float* pooledG = (float*)alloc((size_t)B_ * OC_ * 4);

    hipStream_t s = stream;
    float* out = (float*)d_out;

    // ---- layers 0 & 1 on full graph (all edges valid) ----
    run_conv(nf, ND_, ND_, N0_, ei, nullptr, ef, cp[0], gp[0],
             qb, kb, vb, ebuf, alpha, amaxk, den, agg, skip, hbuf, mu, var, x0, s);
    run_conv(x0, OC_, OC_, N0_, ei, nullptr, ef, cp[1], gp[1],
             qb, kb, vb, ebuf, alpha, amaxk, den, agg, skip, hbuf, mu, var, x1, s);

    // ---- top-k pool #1: 16384 -> 8192 (k=1024 per group of 2048) ----
    score_kernel<<<CDIV(N0_, 256), 256, 0, s>>>(x1, tkw0, score, N0_);
    topk_kernel<<<B_, 1024, 0, s>>>(score, NPG_, NPG_ / 2, perm, vals);
    hipMemsetAsync(ni, 0xFF, (size_t)N0_ * 4, s);  // -1
    ni_scatter_kernel<<<CDIV(8192, 256), 256, 0, s>>>(ni, perm, 8192);
    gather_rows_scale_kernel<<<CDIV(8192 * (OC_ / 4), 256), 256, 0, s>>>(x1, perm, vals, x2, 8192);
    gather_f_kernel<<<CDIV(8192, 256), 256, 0, s>>>(pk,  perm, pkB,  8192);
    gather_f_kernel<<<CDIV(8192, 256), 256, 0, s>>>(pri, perm, priB, 8192);
    gather_i_kernel<<<CDIV(8192, 256), 256, 0, s>>>(batch, perm, batchB, 8192);
    remap_ei_kernel<<<CDIV(E_, 256), 256, 0, s>>>(ei, nullptr, ni, ei1, valid1, E_);
    ic_remap_init_kernel<<<CDIV(MIC_, 256), 256, 0, s>>>(ic, ni, ic0, icv, MIC_);
    em_valid_kernel<<<CDIV(E_, 256), 256, 0, s>>>(ei1, pkB, valid1, E_);

    // ---- layer 2 on pooled graph ----
    run_conv(x2, OC_, OC_, 8192, ei1, valid1, ef, cp[2], gp[2],
             qb, kb, vb, ebuf, alpha, amaxk, den, agg, skip, hbuf, mu, var, x3, s);

    // ---- top-k pool #2: 8192 -> 4096 (k=512 per group of 1024) ----
    score_kernel<<<CDIV(8192, 256), 256, 0, s>>>(x3, tkw1, score, 8192);
    topk_kernel<<<B_, 1024, 0, s>>>(score, NPG_ / 2, NPG_ / 4, perm, vals);
    hipMemsetAsync(ni, 0xFF, (size_t)8192 * 4, s);
    ni_scatter_kernel<<<CDIV(4096, 256), 256, 0, s>>>(ni, perm, 4096);
    gather_rows_scale_kernel<<<CDIV(4096 * (OC_ / 4), 256), 256, 0, s>>>(x3, perm, vals, x4, 4096);
    gather_f_kernel<<<CDIV(4096, 256), 256, 0, s>>>(pkB,  perm, pkC,  4096);
    gather_f_kernel<<<CDIV(4096, 256), 256, 0, s>>>(priB, perm, priC, 4096);
    gather_i_kernel<<<CDIV(4096, 256), 256, 0, s>>>(batchB, perm, batchC, 4096);
    ic_remap_next_kernel<<<CDIV(MIC_, 256), 256, 0, s>>>(ic0, icv, ni, MIC_);

    // ---- heads ----
    nfc_kernel<<<CDIV(4096 * 130, 256), 256, 0, s>>>(x4, pkC, priC, nfc, 4096);

    // iface: rows gathered via ic0, segments = ic[:,1], G = 256
    run_attn_pool(nfc, 130, 130, ic0, MIC_, ic + 1, 2, icv, GIC_, pp[0],
                  gate, pkeys, pden, mproj, pooledI, s);
    head_kernel<<<CDIV(GIC_ * 2, 256), 256, 0, s>>>(pooledI, Wi, bi, out, GIC_, 2, 2, 0);

    // gfold: all nodes, segments = batch, G = 8
    run_attn_pool(nfc, 130, 130, nullptr, 4096, batchC, 1, nullptr, B_, pp[1],
                  gate, pkeys, pden, mproj, pooledF, s);
    head_kernel<<<1, 64, 0, s>>>(pooledF, Wf, bf, out + GIC_ * 2, B_, 3, 6, 0);

    // giface: all nodes, segments = batch, G = 8
    run_attn_pool(nfc, 130, 130, nullptr, 4096, batchC, 1, nullptr, B_, pp[2],
                  gate, pkeys, pden, mproj, pooledG, s);
    head_kernel<<<1, 64, 0, s>>>(pooledG, Wgi, bgi, out + GIC_ * 2, B_, 3, 6, 3);
}
